// GCN_68143951118527
// MI455X (gfx1250) — matrix operand order repaired and checked by reading, not compiled, not run on hardware
//
#include <hip/hip_runtime.h>
#include <hip/hip_bf16.h>
#include <stdint.h>

typedef __attribute__((ext_vector_type(16))) _Float16 v16h;
typedef __attribute__((ext_vector_type(8)))  _Float16 v8h;
typedef __attribute__((ext_vector_type(4)))  _Float16 v4h;
typedef __attribute__((ext_vector_type(8)))  float    v8f;
typedef __attribute__((ext_vector_type(4)))  unsigned int u32x4;
typedef __attribute__((ext_vector_type(8)))  int      i32x8;
typedef __attribute__((ext_vector_type(4)))  int      i32x4;

#define NNODES 50000
#define INC    256
#define OUTC   64
#define MAX_B_HALVES 16384   // 32 KB: full packed W1 panel

// ---------------- elementwise f32 -> f16 (vector x4) ----------------
__global__ __launch_bounds__(256) void k_cvt_f16(const float* __restrict__ in,
                                                 _Float16* __restrict__ out, int n4) {
  int i = blockIdx.x * 256 + threadIdx.x;
  if (i >= n4) return;
  float4 v = ((const float4*)in)[i];
  v4h o;
  o[0] = (_Float16)v.x; o[1] = (_Float16)v.y;
  o[2] = (_Float16)v.z; o[3] = (_Float16)v.w;
  ((v4h*)out)[i] = o;
}

// --------------- pack W (K x 64, f32) into WMMA B-operand layout ---------------
// packed[((kt*4 + nt)*32 + lane)*16 + i] = W[(kt*32 + (lane>>4)*16 + i)*64 + nt*16 + (lane&15)]
__global__ __launch_bounds__(256) void k_pack_w(const float* __restrict__ W,
                                                _Float16* __restrict__ P, int total) {
  int idx = blockIdx.x * 256 + threadIdx.x;
  if (idx >= total) return;
  int i    =  idx        & 15;
  int lane = (idx >> 4)  & 31;
  int nt   = (idx >> 9)  & 3;
  int kt   =  idx >> 11;
  int k = kt * 32 + (lane >> 4) * 16 + i;
  int n = nt * 16 + (lane & 15);
  P[idx] = (_Float16)W[k * OUTC + n];
}

// ---------------- degree: init to 1.0 (self loop), then edge atomics ----------------
__global__ __launch_bounds__(256) void k_fill1(float* __restrict__ deg, int n) {
  int i = blockIdx.x * 256 + threadIdx.x;
  if (i < n) deg[i] = 1.0f;
}

__global__ __launch_bounds__(256) void k_deg(const long long* __restrict__ dst,
                                             float* __restrict__ deg, int E) {
  int e = blockIdx.x * 256 + threadIdx.x;
  if (e < E) atomicAdd(&deg[(int)dst[e]], 1.0f);
}

__global__ __launch_bounds__(256) void k_rsqrt(const float* __restrict__ deg,
                                               float* __restrict__ dinv, int n) {
  int i = blockIdx.x * 256 + threadIdx.x;
  if (i < n) dinv[i] = rsqrtf(deg[i]);
}

// ---- TDM: stage packed B panel (bytes8 * 8 bytes, contiguous) into LDS ----
__device__ __forceinline__ void tdm_stage_b(const _Float16* __restrict__ Bp,
                                            _Float16* __restrict__ Bs,
                                            unsigned int bytes8) {
#if __has_builtin(__builtin_amdgcn_tensor_load_to_lds)
  if (threadIdx.x < 32) {   // one wave issues one DMA (EXEC ignored by TDM)
    unsigned long long ga = (unsigned long long)(uintptr_t)Bp;
    unsigned int lds = (unsigned int)(uintptr_t)Bs;   // low 32 bits = LDS offset
    // D# group 0: count=1 | lds_addr | global_addr[56:0] | type=2
    u32x4 g0;
    g0[0] = 1u;
    g0[1] = lds;
    g0[2] = (unsigned int)(ga & 0xffffffffu);
    g0[3] = (unsigned int)((ga >> 32) & 0x01ffffffu) | (2u << 30);
    // D# group 1: data_size=8B; tensor_dim0=tile_dim0=bytes8; dim1=1; stride0=bytes8
    i32x8 g1;
    g1[0] = (int)(3u << 16);                                   // wg_mask=0, data_size=3 (8B)
    g1[1] = (int)((bytes8 & 0xffffu) << 16);                   // tensor_dim0[15:0]
    g1[2] = (int)(((bytes8 >> 16) & 0xffffu) | (1u << 16));    // tensor_dim0[31:16] | tensor_dim1=1
    g1[3] = (int)((bytes8 & 0xffffu) << 16);                   // tile_dim0 (<=65535)
    g1[4] = 1;                                                 // tile_dim1=1, tile_dim2=0
    g1[5] = (int)bytes8;                                       // tensor_dim0_stride[31:0]
    g1[6] = 0;
    g1[7] = 0;
    i32x4 gz = {0, 0, 0, 0};
#if defined(__clang_major__) && (__clang_major__ >= 23)
    i32x8 gz8 = {0, 0, 0, 0, 0, 0, 0, 0};
    __builtin_amdgcn_tensor_load_to_lds(g0, g1, gz, gz, gz8, 0);
#else
    __builtin_amdgcn_tensor_load_to_lds(g0, g1, gz, gz, 0);
#endif
    __builtin_amdgcn_s_wait_tensorcnt(0);
  }
#else
  // fallback: cooperative copy
  const unsigned long long* src = (const unsigned long long*)Bp;
  unsigned long long* dst = (unsigned long long*)Bs;
  for (unsigned int i = threadIdx.x; i < bytes8; i += 256) dst[i] = src[i];
#endif
  __syncthreads();
}

// ---------------- WMMA GEMM: C[M x 64] = A[M x K](f16) @ Bpacked ----------------
// B panel staged to LDS via TDM; each wave computes two 16-row M tiles (N=64 full).
__global__ __launch_bounds__(256) void k_gemm(const _Float16* __restrict__ A,
                                              const _Float16* __restrict__ Bp,
                                              float* __restrict__ C, int M, int K,
                                              unsigned int bytes8) {
  __shared__ alignas(128) _Float16 Bs[MAX_B_HALVES];
  tdm_stage_b(Bp, Bs, bytes8);

  int wave = threadIdx.x >> 5;
  int lane = threadIdx.x & 31;
  int row0 = (blockIdx.x * 8 + wave) * 32;     // two 16-row tiles per wave
  if (row0 >= M) return;                       // whole-wave exit after barrier
  bool t2 = (row0 + 16) < M;                   // wave-uniform
  int m     = lane & 15;
  int kbase = (lane >> 4) * 16;
  const _Float16* arow0 = A + (size_t)(row0 + m) * K;
  const _Float16* arow1 = arow0 + (size_t)16 * K;

  v8f acc0 = {}, acc1 = {}, acc2 = {}, acc3 = {};
  v8f acc4 = {}, acc5 = {}, acc6 = {}, acc7 = {};
  for (int kk = 0; kk < K; kk += 32) {
    // A fragment: halves 0..15 -> K=kk+kbase+i
    v16h a0 = *(const v16h*)(arow0 + kk + kbase);

    const _Float16* bb = Bs + (size_t)(kk >> 5) * 2048 + (size_t)lane * 16;
    v16h b0 = *(const v16h*)(bb + 0 * 512);
    v16h b1 = *(const v16h*)(bb + 1 * 512);
    v16h b2 = *(const v16h*)(bb + 2 * 512);
    v16h b3 = *(const v16h*)(bb + 3 * 512);

    acc0 = __builtin_amdgcn_wmma_f32_16x16x32_f16(false, a0, false, b0, (short)0, acc0, false, false);
    acc1 = __builtin_amdgcn_wmma_f32_16x16x32_f16(false, a0, false, b1, (short)0, acc1, false, false);
    acc2 = __builtin_amdgcn_wmma_f32_16x16x32_f16(false, a0, false, b2, (short)0, acc2, false, false);
    acc3 = __builtin_amdgcn_wmma_f32_16x16x32_f16(false, a0, false, b3, (short)0, acc3, false, false);

    if (t2) {
      v16h a1 = *(const v16h*)(arow1 + kk + kbase);
      acc4 = __builtin_amdgcn_wmma_f32_16x16x32_f16(false, a1, false, b0, (short)0, acc4, false, false);
      acc5 = __builtin_amdgcn_wmma_f32_16x16x32_f16(false, a1, false, b1, (short)0, acc5, false, false);
      acc6 = __builtin_amdgcn_wmma_f32_16x16x32_f16(false, a1, false, b2, (short)0, acc6, false, false);
      acc7 = __builtin_amdgcn_wmma_f32_16x16x32_f16(false, a1, false, b3, (short)0, acc7, false, false);
    }
  }

  // C/D layout: lanes 0-15: vgpr r -> M=r, N=lane ; lanes 16-31: M=8+r, N=lane-16
  int crow = row0 + (lane >> 4) * 8;
  int ccol = lane & 15;
  #pragma unroll
  for (int r = 0; r < 8; ++r) {
    float* cr = C + (size_t)(crow + r) * OUTC + ccol;
    cr[0]  = acc0[r];
    cr[16] = acc1[r];
    cr[32] = acc2[r];
    cr[48] = acc3[r];
  }
  if (t2) {
    #pragma unroll
    for (int r = 0; r < 8; ++r) {
      float* cr = C + (size_t)(crow + 16 + r) * OUTC + ccol;
      cr[0]  = acc4[r];
      cr[16] = acc5[r];
      cr[32] = acc6[r];
      cr[48] = acc7[r];
    }
  }
}

// ------------- out[i][c] = bias[c] + dinv[i]^2 * h[i][c]  (self-loop + bias init) -------------
__global__ __launch_bounds__(256) void k_init_out(float* __restrict__ out,
                                                  const float* __restrict__ h,
                                                  const float* __restrict__ dinv,
                                                  const float* __restrict__ bias, int total) {
  int idx = blockIdx.x * 256 + threadIdx.x;
  if (idx >= total) return;
  int i = idx >> 6;
  int c = idx & 63;
  float di = dinv[i];
  out[idx] = bias[c] + di * di * h[idx];
}

// ------------- edge scatter: out[dst] += dinv[src]*dinv[dst] * h[src], 1 wave/edge -------------
__global__ __launch_bounds__(256) void k_scatter(const long long* __restrict__ src,
                                                 const long long* __restrict__ dst,
                                                 const float* __restrict__ dinv,
                                                 const float* __restrict__ h,
                                                 float* __restrict__ out, int E) {
  int e = blockIdx.x * 8 + threadIdx.y;
  if (e >= E) return;
  int s = (int)src[e];
  int d = (int)dst[e];
  float nrm = dinv[s] * dinv[d];
  int c = threadIdx.x * 2;
  float2 v = *(const float2*)(h + (size_t)s * OUTC + c);
  float* o = out + (size_t)d * OUTC + c;
  atomicAdd(o + 0, nrm * v.x);
  atomicAdd(o + 1, nrm * v.y);
}

// ------------- fused relu + f32->f16 (vector x4) -------------
__global__ __launch_bounds__(256) void k_relu_cvt(const float* __restrict__ in,
                                                  _Float16* __restrict__ out, int n4) {
  int i = blockIdx.x * 256 + threadIdx.x;
  if (i >= n4) return;
  float4 v = ((const float4*)in)[i];
  v4h o;
  o[0] = (_Float16)fmaxf(v.x, 0.f); o[1] = (_Float16)fmaxf(v.y, 0.f);
  o[2] = (_Float16)fmaxf(v.z, 0.f); o[3] = (_Float16)fmaxf(v.w, 0.f);
  ((v4h*)out)[i] = o;
}

extern "C" void kernel_launch(void* const* d_in, const int* in_sizes, int n_in,
                              void* d_out, int out_size, void* d_ws, size_t ws_size,
                              hipStream_t stream) {
  const float*     x   = (const float*)d_in[0];
  const long long* ei  = (const long long*)d_in[1];
  const float*     W1  = (const float*)d_in[2];
  const float*     b1  = (const float*)d_in[3];
  const float*     W2  = (const float*)d_in[4];
  const float*     b2  = (const float*)d_in[5];
  float*           out = (float*)d_out;

  const int E = in_sizes[1] / 2;
  const long long* src = ei;
  const long long* dst = ei + E;

  // workspace carve-out (256B aligned slices)
  char* ws = (char*)d_ws;
  size_t off = 0;
  auto carve = [&](size_t bytes) {
    void* p = ws + off;
    off = (off + bytes + 255) & ~(size_t)255;
    return p;
  };
  _Float16* xh   = (_Float16*)carve((size_t)NNODES * INC * 2);   // 25.6 MB
  _Float16* w1p  = (_Float16*)carve((size_t)INC * OUTC * 2);     // 32 KB
  _Float16* w2p  = (_Float16*)carve((size_t)OUTC * OUTC * 2);    // 8 KB
  float*    deg  = (float*)carve((size_t)NNODES * 4);
  float*    dinv = (float*)carve((size_t)NNODES * 4);
  float*    h1   = (float*)carve((size_t)NNODES * OUTC * 4);     // 12.8 MB
  float*    out1 = (float*)carve((size_t)NNODES * OUTC * 4);     // 12.8 MB
  _Float16* a1h  = (_Float16*)carve((size_t)NNODES * OUTC * 2);  // 6.4 MB
  float*    h2   = (float*)carve((size_t)NNODES * OUTC * 4);     // 12.8 MB
  (void)ws_size; (void)n_in; (void)out_size;

  const int n4x   = NNODES * INC / 4;
  const int n4h   = NNODES * OUTC / 4;
  const int ntot  = NNODES * OUTC;
  const int mpair = (NNODES / 16 + 1) / 2;       // 1563 pairs of 16-row tiles
  const dim3 sblk(32, 8);

  // stage 0: precision conversion + weight packing + degrees
  k_cvt_f16 <<<(n4x + 255) / 256, 256, 0, stream>>>(x, xh, n4x);
  k_pack_w  <<<(INC * OUTC + 255) / 256, 256, 0, stream>>>(W1, w1p, INC * OUTC);
  k_pack_w  <<<(OUTC * OUTC + 255) / 256, 256, 0, stream>>>(W2, w2p, OUTC * OUTC);
  k_fill1   <<<(NNODES + 255) / 256, 256, 0, stream>>>(deg, NNODES);
  k_deg     <<<(E + 255) / 256, 256, 0, stream>>>(dst, deg, E);
  k_rsqrt   <<<(NNODES + 255) / 256, 256, 0, stream>>>(deg, dinv, NNODES);

  // layer 1  (B panel = INC*OUTC*2 bytes -> /8 for 8-byte TDM units)
  k_gemm    <<<(mpair + 7) / 8, 256, 0, stream>>>(xh, w1p, h1, NNODES, INC,
                                                  (unsigned int)(INC * OUTC * 2 / 8));
  k_init_out<<<(ntot + 255) / 256, 256, 0, stream>>>(out1, h1, dinv, b1, ntot);
  k_scatter <<<(E + 7) / 8, sblk, 0, stream>>>(src, dst, dinv, h1, out1, E);
  k_relu_cvt<<<(n4h + 255) / 256, 256, 0, stream>>>(out1, a1h, n4h);

  // layer 2
  k_gemm    <<<(mpair + 7) / 8, 256, 0, stream>>>(a1h, w2p, h2, NNODES, OUTC,
                                                  (unsigned int)(OUTC * OUTC * 2 / 8));
  k_init_out<<<(ntot + 255) / 256, 256, 0, stream>>>(out, h2, dinv, b2, ntot);
  k_scatter <<<(E + 7) / 8, sblk, 0, stream>>>(src, dst, dinv, h2, out, E);
}
